// SelfAttentionBlock_48885317763631
// MI455X (gfx1250) — matrix operand order repaired and checked by reading, not compiled, not run on hardware
//
#include <hip/hip_runtime.h>

typedef __bf16 bf16;
typedef __attribute__((ext_vector_type(16))) __bf16 v16bf;
typedef __attribute__((ext_vector_type(8)))  float  v8f;
typedef __attribute__((ext_vector_type(4)))  unsigned int u32x4;
typedef __attribute__((ext_vector_type(8)))  int i32x8;
typedef __attribute__((ext_vector_type(4)))  int i32x4;

#define N_TOK 4096
#define C_CH  256
#define BATCH 4

// padded LDS pitches (dword-stride === 4 mod 16 -> ~2-way max bank conflicts)
#define LDW   40    // pitch for 32-wide tiles (32 data + 8 pad)
#define LDK   264   // pitch for 256-wide K^T slab (256 data + 8 pad)

#if defined(__has_builtin)
#if __has_builtin(__builtin_amdgcn_tensor_load_to_lds)
#define HAVE_TDM 1
#endif
#endif
#ifndef HAVE_TDM
#define HAVE_TDM 0
#endif

// ---------- WMMA operand loaders (CDNA5 16x16x32 bf16 layouts, ISA 7.12.2) ----------
__device__ __forceinline__ v16bf wmma_load_a(const bf16* src, int ld) {
  int lane = threadIdx.x & 31;
  const bf16* r = src + (lane & 15) * ld + ((lane >> 4) << 3);
  v16bf a;
#pragma unroll
  for (int t = 0; t < 8; ++t) { a[t] = r[t]; a[t + 8] = r[16 + t]; }
  return a;
}
__device__ __forceinline__ v16bf wmma_load_b(const bf16* srcT, int ld) {
  int lane = threadIdx.x & 31;
  const bf16* r = srcT + (lane & 15) * ld + ((lane >> 4) << 3);
  v16bf b;
#pragma unroll
  for (int t = 0; t < 8; ++t) { b[t] = r[t]; b[t + 8] = r[16 + t]; }
  return b;
}
__device__ __forceinline__ v8f wmma_bf16(v16bf a, v16bf b, v8f c) {
  return __builtin_amdgcn_wmma_f32_16x16x32_bf16(false, a, false, b, (short)0, c, false, false);
}

#if HAVE_TDM
// Issue a 2D TDM tile load (2-byte elements) per ISA D# spec (CDNA5 ISA).
__device__ __forceinline__ void tdm_load_2d(const void* gsrc, unsigned lds_off,
                                            unsigned tile_d0, unsigned tile_d1,
                                            unsigned tensor_d0, unsigned tensor_d1,
                                            unsigned long long stride0,
                                            unsigned pad_interval, unsigned pad_amount,
                                            unsigned pad_en) {
  unsigned long long ga = (unsigned long long)(uintptr_t)gsrc;
  u32x4 g0;
  g0[0] = 1u;                                               // count=1 (valid), user mode
  g0[1] = lds_off;                                          // LDS byte address
  g0[2] = (unsigned)(ga & 0xFFFFFFFFu);                     // global_addr[31:0]
  g0[3] = (unsigned)((ga >> 32) & 0x01FFFFFFu) | (2u << 30);// global_addr[56:32], type=2
  i32x8 g1;
  g1[0] = (int)((1u << 16) | (pad_en << 20) | (pad_interval << 22) | (pad_amount << 25)); // data_size=2B
  g1[1] = (int)((tensor_d0 & 0xFFFFu) << 16);               // tensor_dim0 lo16 (abar=0)
  g1[2] = (int)(((tensor_d0 >> 16) & 0xFFFFu) | ((tensor_d1 & 0xFFFFu) << 16));
  g1[3] = (int)(((tensor_d1 >> 16) & 0xFFFFu) | ((tile_d0 & 0xFFFFu) << 16));
  g1[4] = (int)(tile_d1 & 0xFFFFu);                         // tile_dim2 = 0
  g1[5] = (int)(unsigned)(stride0 & 0xFFFFFFFFull);         // tensor_dim0_stride lo32
  g1[6] = (int)(unsigned)((stride0 >> 32) & 0xFFFFull);     // stride hi16, dim1_stride=0
  g1[7] = 0;
  i32x4 z4 = {0, 0, 0, 0};
#if __clang_major__ >= 23
  i32x8 z8 = {0, 0, 0, 0, 0, 0, 0, 0};
  __builtin_amdgcn_tensor_load_to_lds(g0, g1, z4, z4, z8, 0);
#else
  __builtin_amdgcn_tensor_load_to_lds(g0, g1, z4, z4, 0);
#endif
}
#endif

// ---------------- Kernel 1: BN batch statistics -> folded scale/shift ----------------
__global__ void bn_stats_kernel(const float* __restrict__ x, const float* __restrict__ gamma,
                                const float* __restrict__ beta, float* __restrict__ a_out,
                                float* __restrict__ b_out) {
  __shared__ float s1[256], s2[256];
  int c = blockIdx.x, t = threadIdx.x;
  float sum = 0.f, sq = 0.f;
  for (int b = 0; b < BATCH; ++b) {
    const float* p = x + ((size_t)b * C_CH + c) * N_TOK;
    for (int n = t; n < N_TOK; n += 256) { float v = p[n]; sum += v; sq += v * v; }
  }
  s1[t] = sum; s2[t] = sq;
  __syncthreads();
  for (int off = 128; off > 0; off >>= 1) {
    if (t < off) { s1[t] += s1[t + off]; s2[t] += s2[t + off]; }
    __syncthreads();
  }
  if (t == 0) {
    float inv_n = 1.0f / (float)(BATCH * N_TOK);
    float mean = s1[0] * inv_n;
    float var  = s2[0] * inv_n - mean * mean;
    float a = rsqrtf(var + 1e-5f) * gamma[c];
    a_out[c] = a;
    b_out[c] = beta[c] - mean * a;
  }
}

// ---------------- Kernel 1b: weights f32 -> bf16 (one-shot) ----------------
__global__ void wcvt_kernel(const float* __restrict__ w, bf16* __restrict__ o) {
  int i = blockIdx.x * 1024 + threadIdx.x;
#pragma unroll
  for (int r = 0; r < 4; ++r) o[i + r * 256] = (bf16)w[i + r * 256];
}

// ------------- Kernel 2: normalize + affine + transpose to ht[b][n][c] (bf16) -------------
__global__ void normalize_kernel(const float* __restrict__ x, const float* __restrict__ a_s,
                                 const float* __restrict__ b_s, bf16* __restrict__ ht) {
  __shared__ bf16 tile[64 * 65];
  int tid = threadIdx.x;
  int b = blockIdx.z, c0 = blockIdx.y * 64, n0 = blockIdx.x * 64;
  int col = tid & 63, r4 = tid >> 6;
#pragma unroll
  for (int r = 0; r < 16; ++r) {
    int i = r4 + r * 4;
    int c = c0 + i;
    float v = x[((size_t)b * C_CH + c) * N_TOK + n0 + col];
    tile[i * 65 + col] = (bf16)(v * a_s[c] + b_s[c]);
  }
  __syncthreads();
#pragma unroll
  for (int r = 0; r < 16; ++r) {
    int j = r4 + r * 4;
    ht[((size_t)b * N_TOK + n0 + j) * C_CH + c0 + col] = tile[col * 65 + j];
  }
}

// ---------- shared slab loaders for projection kernels (bf16 weights + ht) ----------
__device__ __forceinline__ void load_w_slab(uint4* sh_w4, const bf16* wb, int cc, int tid) {
  const uint4* src = (const uint4*)(wb + (size_t)tid * C_CH + cc);
  uint4* dst = sh_w4 + tid * (LDW / 8);
#pragma unroll
  for (int q = 0; q < 4; ++q) dst[q] = src[q];
}
__device__ __forceinline__ void load_h_slab(uint4* sh_h4, const bf16* htb, int nbase, int cc, int tid) {
  int r = tid >> 1, half = (tid & 1) * 2;
  const uint4* src = (const uint4*)(htb + (size_t)(nbase + r) * C_CH + cc) + half;
  uint4* dst = sh_h4 + r * (LDW / 8) + half;
  dst[0] = src[0]; dst[1] = src[1];
}

// ------------- Kernel 3a: q/k projection -------------
__global__ __launch_bounds__(256, 1)
void qk_proj_kernel(const bf16* __restrict__ ht, const bf16* __restrict__ wb,
                    const float* __restrict__ bias, bf16* __restrict__ outT) {
  __shared__ uint4 sh_w4[C_CH * (LDW / 8)];
  __shared__ uint4 sh_h4[128 * (LDW / 8)];
  bf16* sh_w = (bf16*)sh_w4;
  bf16* sh_h = (bf16*)sh_h4;
  int tid = threadIdx.x;
  int lane = tid & 31, wv = tid >> 5, hi = lane >> 4, lx = lane & 15;
  int b = blockIdx.y, nbase = blockIdx.x * 128;
  const bf16* htb = ht + (size_t)b * N_TOK * C_CH;
  v8f acc[16];
#pragma unroll
  for (int i = 0; i < 16; ++i)
#pragma unroll
    for (int v = 0; v < 8; ++v) acc[i][v] = 0.f;

  for (int cc = 0; cc < C_CH; cc += 32) {
    load_w_slab(sh_w4, wb, cc, tid);
    load_h_slab(sh_h4, htb, nbase, cc, tid);
    __syncthreads();
    v16bf am = wmma_load_a(sh_h + wv * 16 * LDW, LDW);
    v16bf b0 = wmma_load_b(sh_w, LDW);
    v16bf b1 = wmma_load_b(sh_w + 16 * LDW, LDW);
#pragma unroll
    for (int t = 0; t < 16; t += 2) {              // 2-deep reg-pipelined B operand
      v16bf n0 = wmma_load_b(sh_w + (((t + 2) & 15) * 16) * LDW, LDW);
      v16bf n1 = wmma_load_b(sh_w + (((t + 3) & 15) * 16) * LDW, LDW);
      acc[t]     = wmma_bf16(am, b0, acc[t]);
      acc[t + 1] = wmma_bf16(am, b1, acc[t + 1]);
      b0 = n0; b1 = n1;
    }
    __syncthreads();
  }
  bf16* ob = outT + (size_t)b * N_TOK * C_CH;
#pragma unroll
  for (int t = 0; t < 16; ++t) {
    float bv = bias[t * 16 + lx];
#pragma unroll
    for (int v = 0; v < 8; ++v) {
      int n = nbase + wv * 16 + v + hi * 8;
      ob[(size_t)n * C_CH + t * 16 + lx] = (bf16)(acc[t][v] + bv);
    }
  }
}

// ------------- Kernel 3b: v projection -------------
__global__ __launch_bounds__(256, 1)
void v_proj_kernel(const bf16* __restrict__ ht, const bf16* __restrict__ wb,
                   const float* __restrict__ bias, bf16* __restrict__ vout) {
  __shared__ uint4 sh_w4[C_CH * (LDW / 8)];
  __shared__ uint4 sh_h4[128 * (LDW / 8)];
  bf16* sh_w = (bf16*)sh_w4;
  bf16* sh_h = (bf16*)sh_h4;
  int tid = threadIdx.x;
  int lane = tid & 31, wv = tid >> 5, hi = lane >> 4, lx = lane & 15;
  int b = blockIdx.y, nbase = blockIdx.x * 128;
  const bf16* htb = ht + (size_t)b * N_TOK * C_CH;
  v8f acc[16];
#pragma unroll
  for (int i = 0; i < 16; ++i)
#pragma unroll
    for (int v = 0; v < 8; ++v) acc[i][v] = 0.f;

  for (int cc = 0; cc < C_CH; cc += 32) {
    load_w_slab(sh_w4, wb, cc, tid);
    load_h_slab(sh_h4, htb, nbase, cc, tid);
    __syncthreads();
    v16bf bm = wmma_load_b(sh_h + wv * 16 * LDW, LDW);
    v16bf a0 = wmma_load_a(sh_w, LDW);
    v16bf a1 = wmma_load_a(sh_w + 16 * LDW, LDW);
#pragma unroll
    for (int t = 0; t < 16; t += 2) {
      v16bf n0 = wmma_load_a(sh_w + (((t + 2) & 15) * 16) * LDW, LDW);
      v16bf n1 = wmma_load_a(sh_w + (((t + 3) & 15) * 16) * LDW, LDW);
      acc[t]     = wmma_bf16(a0, bm, acc[t]);
      acc[t + 1] = wmma_bf16(a1, bm, acc[t + 1]);
      a0 = n0; a1 = n1;
    }
    __syncthreads();
  }
  bf16* vbp = vout + (size_t)b * C_CH * N_TOK;
#pragma unroll
  for (int t = 0; t < 16; ++t)
#pragma unroll
    for (int v = 0; v < 8; ++v) {
      int o = t * 16 + v + hi * 8;
      vbp[(size_t)o * N_TOK + nbase + wv * 16 + lx] = (bf16)(acc[t][v] + bias[o]);
    }
}

// ------------- Kernel 4: flash attention; wave = one 16-query tile -------------
__global__ __launch_bounds__(128, 1)
void attention_kernel(const bf16* __restrict__ qt, const bf16* __restrict__ kt,
                      const bf16* __restrict__ vv, bf16* __restrict__ om) {
  __shared__ uint4 sh_kt4[32 * (LDK / 8)];              // 16.9 KB
#if HAVE_TDM
  __shared__ uint4 sh_v4[2 * C_CH * (LDW / 8)];         // 41 KB, TDM double buffer
#else
  __shared__ uint4 sh_v4[C_CH * (LDW / 8)];             // 20.5 KB
#endif
  __shared__ uint4 sh_p4[4 * 16 * (LDW / 8)];           // 5 KB
  bf16* sh_kt = (bf16*)sh_kt4;
  bf16* sh_v  = (bf16*)sh_v4;
  bf16* sh_p  = (bf16*)sh_p4;
  int tid = threadIdx.x;
  int lane = tid & 31, wv = tid >> 5, hi = lane >> 4, lx = lane & 15;
  int b = blockIdx.y;
  size_t bofs = (size_t)b * N_TOK * C_CH;
  int iq = blockIdx.x * 64 + wv * 16;
  const bf16* qtb = qt + bofs;
  const bf16* ktb = kt + bofs;
  const bf16* vb  = vv + bofs;
  const int J = N_TOK / 32;

  v16bf aq[8];
#pragma unroll
  for (int cc = 0; cc < 8; ++cc) aq[cc] = wmma_load_a(qtb + (size_t)iq * C_CH + cc * 32, C_CH);

  v8f acc[16];
#pragma unroll
  for (int i = 0; i < 16; ++i)
#pragma unroll
    for (int v = 0; v < 8; ++v) acc[i][v] = 0.f;
  float mrow[8], lrow[8];
#pragma unroll
  for (int v = 0; v < 8; ++v) { mrow[v] = -3.0e38f; lrow[v] = 0.f; }

  const float SC  = 0.0625f;               // 1/sqrt(C)
  const float L2E = 1.44269504088896f;

  uint4 kreg[8];
  auto load_kregs = [&](int jb) {
    const uint4* ksrc = (const uint4*)(ktb + (size_t)jb * C_CH);
#pragma unroll
    for (int r = 0; r < 8; ++r) kreg[r] = ksrc[tid + r * 128];
  };
#if !HAVE_TDM
  uint4 vreg[8];
  auto load_vregs = [&](int jb) {
#pragma unroll
    for (int rr = 0; rr < 2; ++rr) {
      const uint4* vsrc = (const uint4*)(vb + (size_t)(tid * 2 + rr) * N_TOK + jb);
#pragma unroll
      for (int q = 0; q < 4; ++q) vreg[rr * 4 + q] = vsrc[q];
    }
  };
#endif

#if HAVE_TDM
  // V slab via Tensor Data Mover: tile 32(j) x 256(c), pad 4 dwords every 16 -> pitch LDW
  auto tdm_load_v = [&](int jb, int buf) {
    tdm_load_2d(vb + jb, (unsigned)(uintptr_t)(void*)(sh_v + buf * C_CH * LDW),
                /*tile_d0=*/32, /*tile_d1=*/C_CH,
                /*tensor_d0=*/N_TOK, /*tensor_d1=*/C_CH,
                /*stride0=*/N_TOK, /*pad_interval=*/3, /*pad_amount=*/3, /*pad_en=*/1);
  };
  if (wv == 0) tdm_load_v(0, 0);
#else
  load_vregs(0);
#endif
  load_kregs(0);

  for (int jc = 0; jc < J; ++jc) {
    __syncthreads();                       // previous iteration's LDS consumers done
#pragma unroll
    for (int r = 0; r < 8; ++r) {          // deposit K slab (padded pitch)
      int idx = tid + r * 128;
      sh_kt4[(idx >> 5) * (LDK / 8) + (idx & 31)] = kreg[r];
    }
    if (jc + 1 < J) load_kregs((jc + 1) * 32);           // prefetch next K chunk
#if HAVE_TDM
    if (wv == 0) {
      if (jc + 1 < J) {
        tdm_load_v((jc + 1) * 32, (jc + 1) & 1);         // async DMA next V chunk
        __builtin_amdgcn_s_wait_tensorcnt((short)1);     // older DMA (this chunk) done
      } else {
        __builtin_amdgcn_s_wait_tensorcnt((short)0);
      }
    }
    bf16* svb = sh_v + (jc & 1) * (C_CH * LDW);
#else
#pragma unroll
    for (int rr = 0; rr < 2; ++rr) {
      uint4* dst = sh_v4 + (tid * 2 + rr) * (LDW / 8);
#pragma unroll
      for (int q = 0; q < 4; ++q) dst[q] = vreg[rr * 4 + q];
    }
    if (jc + 1 < J) load_vregs((jc + 1) * 32);
    bf16* svb = sh_v;
#endif
    __syncthreads();

    v8f s0, s1;
#pragma unroll
    for (int v = 0; v < 8; ++v) { s0[v] = 0.f; s1[v] = 0.f; }
    {
      v16bf bk0 = wmma_load_b(sh_kt, LDK);
      v16bf bk1 = wmma_load_b(sh_kt + 16 * LDK, LDK);
#pragma unroll
      for (int cc = 0; cc < 8; ++cc) {     // 2 streams, 1-ahead pipeline
        int nc = (cc + 1) & 7;
        v16bf n0 = wmma_load_b(sh_kt + nc * 32, LDK);
        v16bf n1 = wmma_load_b(sh_kt + 16 * LDK + nc * 32, LDK);
        s0 = wmma_bf16(aq[cc], bk0, s0);
        s1 = wmma_bf16(aq[cc], bk1, s1);
        bk0 = n0; bk1 = n1;
      }
    }

    float p0[8], p1[8], corr[8];
#pragma unroll
    for (int v = 0; v < 8; ++v) {
      float a0 = s0[v] * SC, a1 = s1[v] * SC;
      float mm = fmaxf(a0, a1);
#pragma unroll
      for (int d = 8; d >= 1; d >>= 1) mm = fmaxf(mm, __shfl_xor(mm, d));
      float mn = fmaxf(mrow[v], mm);
      float cr = exp2f((mrow[v] - mn) * L2E);
      corr[v] = cr; mrow[v] = mn;
      float e0 = exp2f((a0 - mn) * L2E);
      float e1 = exp2f((a1 - mn) * L2E);
      p0[v] = e0; p1[v] = e1;
      float rs = e0 + e1;
#pragma unroll
      for (int d = 8; d >= 1; d >>= 1) rs += __shfl_xor(rs, d);
      lrow[v] = lrow[v] * cr + rs;
    }
#pragma unroll
    for (int ct = 0; ct < 16; ++ct)
#pragma unroll
      for (int v = 0; v < 8; ++v) acc[ct][v] *= corr[v];

    // transpose P (C/D layout -> A layout) through wave-private LDS
    bf16* pb = sh_p + wv * 16 * LDW;
#pragma unroll
    for (int v = 0; v < 8; ++v) {
      int row = v + hi * 8;
      pb[row * LDW + lx]      = (bf16)p0[v];
      pb[row * LDW + 16 + lx] = (bf16)p1[v];
    }
    asm volatile("s_wait_dscnt 0" ::: "memory");   // wave-local LDS RAW
    v16bf ap = wmma_load_a(pb, LDW);
    {
      v16bf b0 = wmma_load_b(svb, LDW);
      v16bf b1 = wmma_load_b(svb + 16 * LDW, LDW);
#pragma unroll
      for (int ct = 0; ct < 16; ct += 2) {
        v16bf n0 = wmma_load_b(svb + (((ct + 2) & 15) * 16) * LDW, LDW);
        v16bf n1 = wmma_load_b(svb + (((ct + 3) & 15) * 16) * LDW, LDW);
        acc[ct]     = wmma_bf16(ap, b0, acc[ct]);
        acc[ct + 1] = wmma_bf16(ap, b1, acc[ct + 1]);
        b0 = n0; b1 = n1;
      }
    }
  }

  bf16* ob = om + bofs;
  float inv[8];
#pragma unroll
  for (int v = 0; v < 8; ++v) inv[v] = 1.0f / lrow[v];
#pragma unroll
  for (int ct = 0; ct < 16; ++ct)
#pragma unroll
    for (int v = 0; v < 8; ++v)
      ob[(size_t)(iq + v + hi * 8) * C_CH + ct * 16 + lx] = (bf16)(acc[ct][v] * inv[v]);
}

// ------------- Kernel 5: output projection + residual (f32 out) -------------
__global__ __launch_bounds__(256, 1)
void proj_res_kernel(const bf16* __restrict__ om, const bf16* __restrict__ wb,
                     const float* __restrict__ bias, const float* __restrict__ x,
                     float* __restrict__ out) {
  __shared__ uint4 sh_w4[C_CH * (LDW / 8)];
  __shared__ uint4 sh_h4[128 * (LDW / 8)];
  bf16* sh_w = (bf16*)sh_w4;
  bf16* sh_h = (bf16*)sh_h4;
  int tid = threadIdx.x;
  int lane = tid & 31, wv = tid >> 5, hi = lane >> 4, lx = lane & 15;
  int b = blockIdx.y, nbase = blockIdx.x * 128;
  const bf16* otb = om + (size_t)b * N_TOK * C_CH;
  v8f acc[16];
#pragma unroll
  for (int i = 0; i < 16; ++i)
#pragma unroll
    for (int v = 0; v < 8; ++v) acc[i][v] = 0.f;

  for (int cc = 0; cc < C_CH; cc += 32) {
    load_w_slab(sh_w4, wb, cc, tid);
    load_h_slab(sh_h4, otb, nbase, cc, tid);
    __syncthreads();
    v16bf bm = wmma_load_b(sh_h + wv * 16 * LDW, LDW);
    v16bf a0 = wmma_load_a(sh_w, LDW);
    v16bf a1 = wmma_load_a(sh_w + 16 * LDW, LDW);
#pragma unroll
    for (int t = 0; t < 16; t += 2) {
      v16bf n0 = wmma_load_a(sh_w + (((t + 2) & 15) * 16) * LDW, LDW);
      v16bf n1 = wmma_load_a(sh_w + (((t + 3) & 15) * 16) * LDW, LDW);
      acc[t]     = wmma_bf16(a0, bm, acc[t]);
      acc[t + 1] = wmma_bf16(a1, bm, acc[t + 1]);
      a0 = n0; a1 = n1;
    }
    __syncthreads();
  }
#pragma unroll
  for (int t = 0; t < 16; ++t)
#pragma unroll
    for (int v = 0; v < 8; ++v) {
      int o = t * 16 + v + hi * 8;
      size_t idx = ((size_t)b * C_CH + o) * N_TOK + nbase + wv * 16 + lx;
      out[idx] = acc[t][v] + bias[o] + x[idx];
    }
}

extern "C" void kernel_launch(void* const* d_in, const int* in_sizes, int n_in,
                              void* d_out, int out_size, void* d_ws, size_t ws_size,
                              hipStream_t stream) {
  const float* x     = (const float*)d_in[0];
  const float* gamma = (const float*)d_in[1];
  const float* beta  = (const float*)d_in[2];
  const float* wq    = (const float*)d_in[3];
  const float* bq    = (const float*)d_in[4];
  const float* wk    = (const float*)d_in[5];
  const float* bk    = (const float*)d_in[6];
  const float* wv    = (const float*)d_in[7];
  const float* bv    = (const float*)d_in[8];
  const float* wp    = (const float*)d_in[9];
  const float* bp    = (const float*)d_in[10];
  float* out = (float*)d_out;

  char* ws = (char*)d_ws;
  float* a_s = (float*)ws;
  float* b_s = a_s + 256;
  const size_t M = (size_t)BATCH * N_TOK * C_CH;
  const size_t W = (size_t)C_CH * C_CH;
  bf16* ht = (bf16*)(ws + 4096);
  bf16* qt = ht + M;
  bf16* kt = qt + M;
  bf16* vm = kt + M;
  bf16* om = vm + M;
  bf16* wqb = om + M;
  bf16* wkb = wqb + W;
  bf16* wvb = wkb + W;
  bf16* wpb = wvb + W;

  bn_stats_kernel<<<dim3(C_CH), dim3(256), 0, stream>>>(x, gamma, beta, a_s, b_s);
  wcvt_kernel<<<dim3(64), dim3(256), 0, stream>>>(wq, wqb);
  wcvt_kernel<<<dim3(64), dim3(256), 0, stream>>>(wk, wkb);
  wcvt_kernel<<<dim3(64), dim3(256), 0, stream>>>(wv, wvb);
  wcvt_kernel<<<dim3(64), dim3(256), 0, stream>>>(wp, wpb);
  normalize_kernel<<<dim3(N_TOK / 64, C_CH / 64, BATCH), dim3(256), 0, stream>>>(x, a_s, b_s, ht);
  qk_proj_kernel<<<dim3(N_TOK / 128, BATCH), dim3(256), 0, stream>>>(ht, wqb, bq, qt);
  qk_proj_kernel<<<dim3(N_TOK / 128, BATCH), dim3(256), 0, stream>>>(ht, wkb, bk, kt);
  v_proj_kernel<<<dim3(N_TOK / 128, BATCH), dim3(256), 0, stream>>>(ht, wvb, bv, vm);
  attention_kernel<<<dim3(N_TOK / 64, BATCH), dim3(128), 0, stream>>>(qt, kt, vm, om);
  proj_res_kernel<<<dim3(N_TOK / 128, BATCH), dim3(256), 0, stream>>>(om, wpb, bp, x, out);
}
